// Attention_17222818857539
// MI455X (gfx1250) — compile-verified, matrix operand order checked by reading, not run on hardware
//
#include <hip/hip_runtime.h>
#include <math.h>

// ---------------- problem constants ----------------
static constexpr int Bc = 2, Nc = 4096, Dc = 1024, Hc = 16, NFc = 256;
static constexpr int NCH = 128, CH = 32;            // 128 chunks of 32
static constexpr int Mg = Bc * Nc;                  // 8192 GEMM rows
#define KEPS 1e-4f
#define AEPS 1e-6f

typedef __attribute__((ext_vector_type(16))) __bf16   v16bf;
typedef __attribute__((ext_vector_type(8)))  float    v8f;
typedef __attribute__((ext_vector_type(4)))  unsigned v4u;
typedef __attribute__((ext_vector_type(8)))  int      v8i;
typedef __attribute__((ext_vector_type(4)))  int      v4i;

// ---------------- helpers ----------------
__device__ inline v8f wmma_bf16(v16bf a, v16bf b, v8f c) {
  return __builtin_amdgcn_wmma_f32_16x16x32_bf16(false, a, false, b, (short)0, c,
                                                 false, false);
}

// A-frag from row-major [rows][ld] tile (M on lane&15), or B-frag of Bt stored
// row-major [N][K]. ISA 16-bit layout: lanes 0-15 K={0..7,16..23},
// lanes 16-31 K={8..15,24..31} -> two contiguous 16B runs per lane.
__device__ inline v16bf frag_ld_std(const __bf16* s, int r0, int c0, int ld) {
  const int l = threadIdx.x & 31;
  const int half = l >> 4;
  const __bf16* row = s + (size_t)((l & 15) + r0) * ld + c0 + half * 8;
  v16bf f;
#pragma unroll
  for (int i = 0; i < 8; ++i) { f[i] = row[i]; f[8 + i] = row[16 + i]; }
  return f;
}

// Fragment whose 16-index is the minor dim of a [K][ld] row-major source (bf16).
__device__ inline v16bf frag_ld_kmaj_bf(const __bf16* s, int k0, int n0, int ld) {
  const int l = threadIdx.x & 31;
  const int half = l >> 4;
  const int n = n0 + (l & 15);
  v16bf f;
#pragma unroll
  for (int e = 0; e < 16; ++e) {
    int k = ((e < 8) ? e : e + 8) + half * 8 + k0;
    f[e] = s[(size_t)k * ld + n];
  }
  return f;
}

// Same, f32 source converted at fragment build.
__device__ inline v16bf frag_ld_kmaj_f32(const float* s, int k0, int n0, int ld) {
  const int l = threadIdx.x & 31;
  const int half = l >> 4;
  const int n = n0 + (l & 15);
  v16bf f;
#pragma unroll
  for (int e = 0; e < 16; ++e) {
    int k = ((e < 8) ? e : e + 8) + half * 8 + k0;
    f[e] = (__bf16)s[(size_t)k * ld + n];
  }
  return f;
}

// Monotone float<->uint encoding: unsigned atomicMax == float max (deterministic).
__device__ inline unsigned fenc(float x) {
  unsigned u = __float_as_uint(x);
  return (u & 0x80000000u) ? ~u : (u | 0x80000000u);
}
__device__ inline float fdec(unsigned u) {
  u = (u & 0x80000000u) ? (u & 0x7FFFFFFFu) : ~u;
  return __uint_as_float(u);
}

// TDM: 2-D f32 tile (tile_d1 rows x tile_d0 elems, row stride stride0 elems)
// from global into LDS at byte offset lds_off. D# per CDNA5 ISA ch.8:
//   group0: [1:0]=count=1, [63:32]=lds_addr, [120:64]=global_addr, [127:126]=type=2
//   group1: [17:16]=data_size(2->4B), [79:48]=tensor_dim0, [111:80]=tensor_dim1,
//           [127:112]=tile_dim0, [143:128]=tile_dim1, [207:160]=tensor_dim0_stride
// Groups 2/3 unused for 2-D tensors (zero-filled) — clang-23 6-arg builtin form.
__device__ inline void tdm_load_2d_f32(const void* gptr, unsigned lds_off,
                                       unsigned td0, unsigned td1,
                                       unsigned tile_d0, unsigned tile_d1,
                                       unsigned stride0) {
  unsigned long long ga = (unsigned long long)(uintptr_t)gptr;
  v4u g0;
  g0[0] = 1u;                                          // count=1 (user descriptor)
  g0[1] = lds_off;                                     // lds_addr (bytes)
  g0[2] = (unsigned)(ga & 0xFFFFFFFFu);
  g0[3] = (unsigned)((ga >> 32) & 0x01FFFFFFu) | (2u << 30);  // addr[56:32] | type=2
  v8i g1;
  g1[0] = (int)(2u << 16);                             // wg_mask=0, data_size=4B
  g1[1] = (int)((td0 & 0xFFFFu) << 16);                // tensor_dim0[15:0]
  g1[2] = (int)((td0 >> 16) | ((td1 & 0xFFFFu) << 16));
  g1[3] = (int)((td1 >> 16) | (tile_d0 << 16));        // tile_dim0
  g1[4] = (int)(tile_d1 & 0xFFFFu);                    // tile_dim1, tile_dim2=0
  g1[5] = (int)stride0;                                // dim0_stride[31:0]
  g1[6] = 0;                                           // stride hi / dim1_stride
  g1[7] = 0;
  v4i z4 = {0, 0, 0, 0};
  v8i z8 = {0, 0, 0, 0, 0, 0, 0, 0};
  __builtin_amdgcn_tensor_load_to_lds(g0, g1, z4, z4, z8, 0);
}

// ---------------- Kernel 1/6: C[8192x1024] = A @ W^T (+bias) ----------------
template <bool HAS_BIAS>
__global__ __launch_bounds__(256) void gemm_nt_kernel(
    const float* __restrict__ A, const float* __restrict__ W,
    const float* __restrict__ bias, float* __restrict__ C) {
  constexpr int K = 1024, O = 1024;
  __shared__ __bf16 As[128 * 32];
  __shared__ __bf16 Ws[128 * 32];
  const int tid = threadIdx.x, wid = tid >> 5, lane = tid & 31;
  const int ln = lane & 15, hb = lane >> 4;
  const int m0 = blockIdx.y * 128, o0 = blockIdx.x * 128;
  const int wm = wid & 3, wn = wid >> 2;
  v8f acc[2][4];
#pragma unroll
  for (int mi = 0; mi < 2; ++mi)
#pragma unroll
    for (int ni = 0; ni < 4; ++ni) acc[mi][ni] = v8f{};

  for (int k0 = 0; k0 < K; k0 += 32) {
#pragma unroll
    for (int i = 0; i < 4; ++i) {          // 1024 float4 per tile / 256 threads
      int f4 = tid + i * 256;
      int r = f4 >> 3, c = (f4 & 7) * 4;
      float4 av = *(const float4*)(A + (size_t)(m0 + r) * K + k0 + c);
      As[r * 32 + c + 0] = (__bf16)av.x; As[r * 32 + c + 1] = (__bf16)av.y;
      As[r * 32 + c + 2] = (__bf16)av.z; As[r * 32 + c + 3] = (__bf16)av.w;
      float4 wv = *(const float4*)(W + (size_t)(o0 + r) * K + k0 + c);
      Ws[r * 32 + c + 0] = (__bf16)wv.x; Ws[r * 32 + c + 1] = (__bf16)wv.y;
      Ws[r * 32 + c + 2] = (__bf16)wv.z; Ws[r * 32 + c + 3] = (__bf16)wv.w;
    }
    if (k0 + 32 < K) {                     // prefetch next K-tile into caches
      int r = tid >> 3, c = (tid & 7) * 4;
      __builtin_prefetch(A + (size_t)(m0 + r) * K + k0 + 32 + c);
      __builtin_prefetch(W + (size_t)(o0 + r) * K + k0 + 32 + c);
    }
    __syncthreads();
    v16bf af[2], bf[4];
#pragma unroll
    for (int mi = 0; mi < 2; ++mi) af[mi] = frag_ld_std(As, wm * 32 + mi * 16, 0, 32);
#pragma unroll
    for (int ni = 0; ni < 4; ++ni) bf[ni] = frag_ld_std(Ws, wn * 64 + ni * 16, 0, 32);
#pragma unroll
    for (int mi = 0; mi < 2; ++mi)
#pragma unroll
      for (int ni = 0; ni < 4; ++ni) acc[mi][ni] = wmma_bf16(af[mi], bf[ni], acc[mi][ni]);
    __syncthreads();
  }
  float* cb = C + (size_t)(m0 + wm * 32 + hb * 8) * O + o0 + wn * 64 + ln;
#pragma unroll
  for (int mi = 0; mi < 2; ++mi)
#pragma unroll
    for (int ni = 0; ni < 4; ++ni) {
      float bv = 0.f;
      if (HAS_BIAS) bv = bias[o0 + wn * 64 + ni * 16 + ln];
#pragma unroll
      for (int r = 0; r < 8; ++r)
        cb[(size_t)(mi * 16 + r) * O + ni * 16] = acc[mi][ni][r] + bv;
    }
}

// ---------------- Kernel 2-3/6: feature map GEMM (+exp epilogue) ----------------
__global__ __launch_bounds__(256) void feat_kernel(
    const float* __restrict__ qk, const float* __restrict__ proj,
    float* __restrict__ outp, float* __restrict__ diag_out,
    unsigned* __restrict__ gmax_u, int is_query) {
  __shared__ __bf16 Ps[256 * 64];
  __shared__ __bf16 As[64 * 64];
  __shared__ float ssq[64 * 4];
  __shared__ unsigned rmax_u[64];
  __shared__ unsigned bmax_u;
  const int tid = threadIdx.x, wid = tid >> 5, lane = tid & 31;
  const int ln = lane & 15, hb = lane >> 4;
  const int wm = wid & 3, wn = wid >> 2;
  const int n0 = blockIdx.x * 64;
  const int bh = blockIdx.y, b = bh / Hc, h = bh % Hc;
  const float dn = 0.3535533905932738f;   // 64^-0.25
  const float ratio = 0.0625f;            // 256^-0.5
  const float* Abase = qk + (size_t)b * Nc * Dc + h * 64;

#pragma unroll
  for (int i = 0; i < 16; ++i) {          // proj: 4096 float4
    int f4 = tid + i * 256; int r = f4 >> 4, c = (f4 & 15) * 4;
    float4 v = *(const float4*)(proj + (size_t)r * 64 + c);
    Ps[r * 64 + c + 0] = (__bf16)v.x; Ps[r * 64 + c + 1] = (__bf16)v.y;
    Ps[r * 64 + c + 2] = (__bf16)v.z; Ps[r * 64 + c + 3] = (__bf16)v.w;
  }
#pragma unroll
  for (int i = 0; i < 4; ++i) {           // A: 1024 float4
    int f4 = tid + i * 256; int r = f4 >> 4, c = (f4 & 15) * 4;
    float4 v = *(const float4*)(Abase + (size_t)(n0 + r) * Dc + c);
    As[r * 64 + c + 0] = (__bf16)v.x; As[r * 64 + c + 1] = (__bf16)v.y;
    As[r * 64 + c + 2] = (__bf16)v.z; As[r * 64 + c + 3] = (__bf16)v.w;
  }
  if (tid < 64) rmax_u[tid] = 0u;
  if (tid == 0) bmax_u = 0u;
  __syncthreads();

  { // per-row sum of squares (4 threads/row)
    int row = tid >> 2, q = tid & 3;
    float s = 0.f;
#pragma unroll
    for (int j = 0; j < 16; ++j) { float x = (float)As[row * 64 + q * 16 + j]; s += x * x; }
    ssq[row * 4 + q] = s;
  }

  v8f acc[8];
#pragma unroll
  for (int ni = 0; ni < 8; ++ni) acc[ni] = v8f{};
#pragma unroll
  for (int ks = 0; ks < 64; ks += 32) {
    v16bf a = frag_ld_std(As, wm * 16, ks, 64);
#pragma unroll
    for (int ni = 0; ni < 8; ++ni) {
      v16bf bb = frag_ld_std(Ps, wn * 128 + ni * 16, ks, 64);
      acc[ni] = wmma_bf16(a, bb, acc[ni]);
    }
  }

  if (is_query) {
#pragma unroll
    for (int r = 0; r < 8; ++r) {
      float pm = dn * acc[0][r];
#pragma unroll
      for (int ni = 1; ni < 8; ++ni) pm = fmaxf(pm, dn * acc[ni][r]);
      atomicMax(&rmax_u[wm * 16 + r + hb * 8], fenc(pm));
    }
  } else {
    float pm = dn * acc[0][0];
#pragma unroll
    for (int ni = 0; ni < 8; ++ni)
#pragma unroll
      for (int r = 0; r < 8; ++r) pm = fmaxf(pm, dn * acc[ni][r]);
    atomicMax(&bmax_u, fenc(pm));
  }
  __syncthreads();

  const float dscale = 0.5f * dn * dn;
  float* orow = outp + ((size_t)bh * Nc + n0) * NFc;
#pragma unroll
  for (int ni = 0; ni < 8; ++ni)
#pragma unroll
    for (int r = 0; r < 8; ++r) {
      int rl = wm * 16 + r + hb * 8;
      int col = wn * 128 + ni * 16 + ln;
      float ddv = dn * acc[ni][r];
      float dg = dscale * (ssq[rl * 4 + 0] + ssq[rl * 4 + 1] + ssq[rl * 4 + 2] + ssq[rl * 4 + 3]);
      if (is_query) {
        float m = fdec(rmax_u[rl]);
        orow[(size_t)rl * NFc + col] = ratio * (expf(ddv - dg - m) + KEPS);
      } else {
        orow[(size_t)rl * NFc + col] = ddv;
      }
    }
  if (!is_query) {
    if (tid < 64)
      diag_out[(size_t)bh * Nc + n0 + tid] =
          dscale * (ssq[tid * 4 + 0] + ssq[tid * 4 + 1] + ssq[tid * 4 + 2] + ssq[tid * 4 + 3]);
    if (tid == 0) atomicMax(gmax_u, bmax_u);
  }
}

// ---------------- Kernel 4/6: key exp epilogue with global max ----------------
__global__ void key_exp_kernel(float* __restrict__ dd, const float* __restrict__ diag,
                               const unsigned* __restrict__ gmax_u, size_t total) {
  const float m = fdec(*gmax_u);
  for (size_t i = (size_t)blockIdx.x * blockDim.x + threadIdx.x; i < total;
       i += (size_t)gridDim.x * blockDim.x) {
    size_t row = i >> 8;
    dd[i] = 0.0625f * (expf(dd[i] - diag[row] - m) + KEPS);
  }
}

// ---------------- Kernel 5/6: causal chunked scan (one WG per (b,h)) ----------------
// ctx[256x64] f32 + kcum[256] resident in LDS across 128 chunks; chunk staging via TDM.
__global__ __launch_bounds__(256) void scan_kernel(
    const float* __restrict__ qp, const float* __restrict__ kp,
    const float* __restrict__ v, float* __restrict__ out) {
  extern __shared__ char smem[];
  float* ctx   = (float*)smem;              // 256*64
  float* kcum  = ctx + 256 * 64;            // 256
  float* Am    = kcum + 256;                // 32*32
  float* dinv  = Am + 32 * 32;              // 32
  float* dbase = dinv + 32;                 // 32*8
  float* qstg  = dbase + 256;               // 32*256 (TDM dest)
  float* kstg  = qstg + 32 * 256;           // 32*256 (TDM dest)
  float* vstg  = kstg + 32 * 256;           // 32*64  (TDM dest)
  __bf16* qq   = (__bf16*)(vstg + 32 * 64); // 32*256
  __bf16* kk   = qq + 32 * 256;             // 32*256
  __bf16* vvs  = kk + 32 * 256;             // 32*64
  __bf16* Ab   = vvs + 32 * 64;             // 32*32

  const int tid = threadIdx.x, wid = tid >> 5, lane = tid & 31;
  const int ln = lane & 15, hb = lane >> 4;
  const int bh = blockIdx.x, b = bh >> 4, h = bh & 15;
  const float* qpb = qp + (size_t)bh * Nc * NFc;
  const float* kpb = kp + (size_t)bh * Nc * NFc;
  const float* vb  = v  + (size_t)b * Nc * Dc + h * 64;
  float* ob        = out + (size_t)b * Nc * Dc + h * 64;

  const unsigned qstg_off = (unsigned)(uintptr_t)(void*)qstg;
  const unsigned kstg_off = (unsigned)(uintptr_t)(void*)kstg;
  const unsigned vstg_off = (unsigned)(uintptr_t)(void*)vstg;

  for (int i = tid; i < 256 * 64; i += 256) ctx[i] = 0.f;
  if (tid < 256) kcum[tid] = 0.f;
  __syncthreads();

  for (int t = 0; t < NCH; ++t) {
    const int n0 = t * CH;
    if (wid == 0) {                          // one wave drives the TDM
      tdm_load_2d_f32(qpb + (size_t)n0 * NFc, qstg_off, 256, 32, 256, 32, 256);
      tdm_load_2d_f32(kpb + (size_t)n0 * NFc, kstg_off, 256, 32, 256, 32, 256);
      tdm_load_2d_f32(vb + (size_t)n0 * Dc, vstg_off, 64, 32, 64, 32, Dc);
      __builtin_amdgcn_s_wait_tensorcnt(0);
    }
    __syncthreads();                         // staging visible to all waves

#pragma unroll 4
    for (int i = 0; i < 32; ++i) {           // f32 staging -> bf16 operands
      int e = tid + i * 256;
      qq[e] = (__bf16)qstg[e];
      kk[e] = (__bf16)kstg[e];
    }
#pragma unroll
    for (int i = 0; i < 8; ++i) {
      int e = tid + i * 256;
      vvs[e] = (__bf16)vstg[e];
    }
    { // d_base[i] = qq_i . (kcum0 + eps)   (8 threads per row)
      int row = tid >> 3, seg = tid & 7;
      const float* qrow = qstg + row * 256 + seg * 32;
      float s = 0.f;
#pragma unroll
      for (int j = 0; j < 32; ++j) s += qrow[j] * (kcum[seg * 32 + j] + AEPS);
      dbase[row * 8 + seg] = s;
    }
    __syncthreads();

    if (wid < 4) {                           // A = qq @ kk^T  (32x32, K=256)
      const int mi = wid & 1, ni = wid >> 1;
      v8f acc = v8f{};
#pragma unroll
      for (int ks = 0; ks < 256; ks += 32) {
        v16bf a  = frag_ld_std(qq, mi * 16, ks, 256);
        v16bf bb = frag_ld_std(kk, ni * 16, ks, 256);
        acc = wmma_bf16(a, bb, acc);
      }
#pragma unroll
      for (int r = 0; r < 8; ++r)
        Am[(mi * 16 + r + hb * 8) * 32 + ni * 16 + ln] = acc[r];
    }
    __syncthreads();

#pragma unroll
    for (int i = 0; i < 4; ++i) {            // causal mask (inclusive) -> bf16
      int e = tid * 4 + i; int ii = e >> 5, jj = e & 31;
      float mv = (jj <= ii) ? Am[e] : 0.f;
      Ab[e] = (__bf16)mv;
    }
    __syncthreads();

    if (tid < 32) {                          // d_i and reciprocal
      float s = 0.f;
#pragma unroll
      for (int q = 0; q < 8; ++q) s += dbase[tid * 8 + q];
      for (int j = 0; j <= tid; ++j) s += Am[tid * 32 + j];
      dinv[tid] = 1.0f / s;
    }
    __syncthreads();

    {                                        // out = (qq@ctx0 + tril(A)@vv) * dinv
      const int mi = wid & 1, ni = wid >> 1;
      v8f acc = v8f{};
#pragma unroll
      for (int ks = 0; ks < 256; ks += 32) {
        v16bf a  = frag_ld_std(qq, mi * 16, ks, 256);
        v16bf bb = frag_ld_kmaj_f32(ctx, ks, ni * 16, 64);
        acc = wmma_bf16(a, bb, acc);
      }
      v16bf a2  = frag_ld_std(Ab, mi * 16, 0, 32);
      v16bf bb2 = frag_ld_kmaj_bf(vvs, 0, ni * 16, 64);
      acc = wmma_bf16(a2, bb2, acc);
#pragma unroll
      for (int r = 0; r < 8; ++r) {
        int rl = mi * 16 + r + hb * 8;
        ob[(size_t)(n0 + rl) * Dc + ni * 16 + ln] = acc[r] * dinv[rl];
      }
    }
    __syncthreads();                         // all ctx0 reads done

#pragma unroll
    for (int q = 0; q < 8; ++q) {            // ctx += kk^T @ vv (64 tiles / 8 waves)
      int tt = wid + q * 8; int fi = tt & 15, ei = tt >> 4;
      v8f c;
#pragma unroll
      for (int r = 0; r < 8; ++r)
        c[r] = ctx[(fi * 16 + r + hb * 8) * 64 + ei * 16 + ln];
      v16bf a  = frag_ld_kmaj_bf(kk, 0, fi * 16, 256);
      v16bf bb = frag_ld_kmaj_bf(vvs, 0, ei * 16, 64);
      c = wmma_bf16(a, bb, c);
#pragma unroll
      for (int r = 0; r < 8; ++r)
        ctx[(fi * 16 + r + hb * 8) * 64 + ei * 16 + ln] = c[r];
    }
    {                                        // kcum += colsum(kk)
      float s = 0.f;
#pragma unroll
      for (int j = 0; j < 32; ++j) s += (float)kk[j * 256 + tid];
      kcum[tid] += s;
    }
    __syncthreads();
  }
}

// ---------------- launch ----------------
extern "C" void kernel_launch(void* const* d_in, const int* in_sizes, int n_in,
                              void* d_out, int out_size, void* d_ws, size_t ws_size,
                              hipStream_t stream) {
  (void)in_sizes; (void)n_in; (void)out_size; (void)ws_size;
  const float* x    = (const float*)d_in[0];
  const float* wq   = (const float*)d_in[1];
  const float* wk   = (const float*)d_in[2];
  const float* wv   = (const float*)d_in[3];
  const float* wo   = (const float*)d_in[4];
  const float* bo   = (const float*)d_in[5];
  const float* proj = (const float*)d_in[6];
  float* outp = (float*)d_out;

  const size_t QS = (size_t)Bc * Nc * Dc;          // 8.39M floats
  const size_t FS = (size_t)Bc * Hc * Nc * NFc;    // 33.55M floats
  char* ws = (char*)d_ws;
  float* qb   = (float*)ws;                        ws += QS * 4;
  float* kb   = (float*)ws;                        ws += QS * 4;
  float* vb   = (float*)ws;                        ws += QS * 4;
  float* ao   = (float*)ws;                        ws += QS * 4;
  float* qp   = (float*)ws;                        ws += FS * 4;
  float* kp   = (float*)ws;                        ws += FS * 4;   // dd_k, exp'd in place
  float* dgk  = (float*)ws;                        ws += (size_t)Bc * Hc * Nc * 4;
  unsigned* gmax_u = (unsigned*)ws;

  (void)hipMemsetAsync(gmax_u, 0, sizeof(unsigned), stream);   // encoded -inf

  dim3 ggrid(Dc / 128, Mg / 128);        // 8 x 64
  gemm_nt_kernel<false><<<ggrid, 256, 0, stream>>>(x, wq, nullptr, qb);
  gemm_nt_kernel<false><<<ggrid, 256, 0, stream>>>(x, wk, nullptr, kb);
  gemm_nt_kernel<false><<<ggrid, 256, 0, stream>>>(x, wv, nullptr, vb);

  dim3 fgrid(Nc / 64, Bc * Hc);          // 64 x 32
  feat_kernel<<<fgrid, 256, 0, stream>>>(qb, proj, qp, nullptr, nullptr, 1);
  feat_kernel<<<fgrid, 256, 0, stream>>>(kb, proj, kp, dgk, gmax_u, 0);
  key_exp_kernel<<<8192, 256, 0, stream>>>(kp, dgk, gmax_u, FS);

  const size_t scan_lds =
      (256 * 64 + 256 + 32 * 32 + 32 + 256 + 32 * 256 * 2 + 32 * 64) * sizeof(float) +
      (32 * 256 * 2 + 32 * 64 + 32 * 32) * sizeof(__bf16);
  scan_kernel<<<Bc * Hc, 256, scan_lds, stream>>>(qp, kp, vb, ao);

  gemm_nt_kernel<true><<<ggrid, 256, 0, stream>>>(ao, wo, bo, outp);
}
